// Seq2Seq_with_attention_modified_56719338111705
// MI455X (gfx1250) — matrix operand order repaired and checked by reading, not compiled, not run on hardware
//
#include <hip/hip_runtime.h>
#include <hip/hip_bf16.h>

// Problem constants (from reference): B=512, WIN=128, D=256, H=512, O=64, P=64
#define B_   512
#define WIN_ 128
#define D_   256
#define H_   512
#define O_   64
#define P_   64
#define KIN_ (O_ + D_ + H_)   // 832 = LSTM combined input width [y | ctx | h]

typedef __attribute__((ext_vector_type(16))) __bf16 v16bf;
typedef __attribute__((ext_vector_type(8)))  float  v8f;

#if __has_builtin(__builtin_amdgcn_global_load_async_to_lds_b128) && __has_builtin(__builtin_amdgcn_s_wait_asynccnt)
#define HAVE_ASYNC 1
// Exact parameter types per hipcc diagnostic:
//   arg0: int __vector(4) in AS(1)  (global source)
//   arg1: int __vector(4) in AS(3)  (LDS destination)
typedef int async_vec __attribute__((vector_size(16)));
typedef __attribute__((address_space(1))) async_vec* g_async_p;
typedef __attribute__((address_space(3))) async_vec* l_async_p;
#endif

__device__ __forceinline__ unsigned pk_bf16(float a, float b) {
  unsigned short lo = __builtin_bit_cast(unsigned short, (__bf16)a);
  unsigned short hi = __builtin_bit_cast(unsigned short, (__bf16)b);
  return (unsigned)lo | ((unsigned)hi << 16);
}

__device__ __forceinline__ float sigmoidf_(float x) { return 1.0f / (1.0f + expf(-x)); }

union FragBF { v16bf v; unsigned u[8]; };

#define LDSP 40   // padded LDS row stride (bf16 elems): 80B rows -> conflict-free 16-row gathers

// ---------------------------------------------------------------------------
// Generic bf16 WMMA GEMM:  C[M,N] = A[M,K] (f32, row-major) * Bw[N,K]^T (bf16,
// row-major over N) + bias[N].  Block tile 128(M) x 64(N), K-step 32.
// 256 threads = 8 waves arranged 4(M) x 2(N); each wave -> 32x32 via 2x2 WMMA.
// Optional second output C2 (dual store). Batched via blockIdx.z + strides.
// ---------------------------------------------------------------------------
__global__ __launch_bounds__(256)
void gemm_bf16_tn(const float* __restrict__ A, int lda, long long strideA,
                  const __bf16* __restrict__ Bw, int ldb,
                  float* __restrict__ C, int ldc, long long strideC,
                  float* __restrict__ C2, int ldc2,
                  const float* __restrict__ bias, int K)
{
  __shared__ __bf16 As[128 * LDSP];
  __shared__ __bf16 Bs[64 * LDSP];

  const int bz = blockIdx.z;
  A += (long long)bz * strideA;
  C += (long long)bz * strideC;

  const int row0 = blockIdx.y * 128;
  const int col0 = blockIdx.x * 64;
  const int t    = threadIdx.x;
  const int lane = t & 31;
  const int wave = t >> 5;
  const int m16  = lane & 15;
  const int khalf = lane >> 4;
  const int waveM = (wave & 3) * 32;   // 4 waves along M
  const int waveN = (wave >> 2) * 32;  // 2 waves along N

  const v8f vzero = {0.f,0.f,0.f,0.f,0.f,0.f,0.f,0.f};
  v8f acc[2][2];
  acc[0][0] = vzero; acc[0][1] = vzero; acc[1][0] = vzero; acc[1][1] = vzero;

  // A loader: thread -> row ar (0..127), 16 consecutive floats at col ah
  const int ar = t >> 1;
  const int ah = (t & 1) * 16;
  const float* Abase = A + (long long)(row0 + ar) * lda + ah;

  // B loader: thread -> row brow (0..63), 8 bf16 (16B) chunk at col bco
  const int brow = t >> 2;
  const int bco  = (t & 3) * 8;
  const __bf16* Bbase = Bw + (long long)(col0 + brow) * ldb + bco;

  for (int k0 = 0; k0 < K; k0 += 32) {
    __syncthreads();   // protect LDS from previous iteration's readers

    if (k0 + 32 < K) { // prefetch next tiles -> global_prefetch_b8
      __builtin_prefetch(Abase + k0 + 32, 0, 3);
      __builtin_prefetch(Bbase + k0 + 32, 0, 3);
    }

    // ---- stage A tile (f32 -> bf16) ----
    const float4* ap = (const float4*)(Abase + k0);
    float4 f0 = ap[0], f1 = ap[1], f2 = ap[2], f3 = ap[3];
    unsigned* arow = (unsigned*)(As + ar * LDSP + ah);
    arow[0] = pk_bf16(f0.x, f0.y); arow[1] = pk_bf16(f0.z, f0.w);
    arow[2] = pk_bf16(f1.x, f1.y); arow[3] = pk_bf16(f1.z, f1.w);
    arow[4] = pk_bf16(f2.x, f2.y); arow[5] = pk_bf16(f2.z, f2.w);
    arow[6] = pk_bf16(f3.x, f3.y); arow[7] = pk_bf16(f3.z, f3.w);

    // ---- stage B tile (already bf16): async DMA to LDS when available ----
#if defined(HAVE_ASYNC)
    __builtin_amdgcn_global_load_async_to_lds_b128(
        (g_async_p)(Bbase + k0),
        (l_async_p)(Bs + brow * LDSP + bco),
        0, 0);
    __builtin_amdgcn_s_wait_asynccnt(0);
#else
    uint4 bv = *(const uint4*)(Bbase + k0);
    *(uint4*)(Bs + brow * LDSP + bco) = bv;
#endif

    __syncthreads();

    // ---- gather fragments per documented wave32 layouts ----
    FragBF afrag[2], bfrag[2];
#pragma unroll
    for (int v = 0; v < 8; ++v) {
      const int ka = 8 * khalf + ((v < 4) ? (2 * v) : (8 + 2 * v));  // A 16x32 bf16 layout
      const int kb = 8 * khalf + ((v < 4) ? (2 * v) : (8 + 2 * v));  // B 32x16 bf16 layout
      afrag[0].u[v] = *(const unsigned*)(As + (waveM + m16) * LDSP + ka);
      afrag[1].u[v] = *(const unsigned*)(As + (waveM + 16 + m16) * LDSP + ka);
      bfrag[0].u[v] = *(const unsigned*)(Bs + (waveN + m16) * LDSP + kb);
      bfrag[1].u[v] = *(const unsigned*)(Bs + (waveN + 16 + m16) * LDSP + kb);
    }

#pragma unroll
    for (int i = 0; i < 2; ++i)
#pragma unroll
      for (int j = 0; j < 2; ++j)
        acc[i][j] = __builtin_amdgcn_wmma_f32_16x16x32_bf16(
            false, afrag[i].v, false, bfrag[j].v, (short)0, acc[i][j], false, false);
  }

  // ---- epilogue: C/D 16x16 f32 layout: lane n = lane&15, row = r + 8*khalf ----
#pragma unroll
  for (int i = 0; i < 2; ++i) {
#pragma unroll
    for (int j = 0; j < 2; ++j) {
      const int mbase = row0 + waveM + 16 * i + 8 * khalf;
      const int nn    = col0 + waveN + 16 * j + m16;
      const float bvl = bias ? bias[nn] : 0.0f;
#pragma unroll
      for (int r = 0; r < 8; ++r) {
        const float val = acc[i][j][r] + bvl;
        C[(long long)(mbase + r) * ldc + nn] = val;
        if (C2) C2[(long long)(mbase + r) * ldc2 + nn] = val;
      }
    }
  }
}

// ---------------------------------------------------------------------------
// One-time preprocessing kernels
// ---------------------------------------------------------------------------
__global__ void cvt_bf16(const float* __restrict__ in, __bf16* __restrict__ out, int n) {
  int i = blockIdx.x * blockDim.x + threadIdx.x;
  if (i < n) out[i] = (__bf16)in[i];
}

__global__ void build_wcomb(const float* __restrict__ Wih, const float* __restrict__ Whh,
                            __bf16* __restrict__ out) {
  int i = blockIdx.x * blockDim.x + threadIdx.x;
  if (i >= 4 * H_ * KIN_) return;
  int n = i / KIN_, k = i % KIN_;
  float v = (k < O_ + D_) ? Wih[n * (O_ + D_) + k] : Whh[n * H_ + (k - (O_ + D_))];
  out[i] = (__bf16)v;
}

__global__ void vec_add(const float* __restrict__ a, const float* __restrict__ b,
                        float* __restrict__ o, int n) {
  int i = blockIdx.x * blockDim.x + threadIdx.x;
  if (i < n) o[i] = a[i] + b[i];
}

// h0 = tanh(x_last @ Ws^T + bs); c0 = sigmoid(x_last @ Wc^T + bc)
// Writes hc = [h | c], seeds inp: y-cols <- 0, h-cols <- h0.
__global__ void init_state(const float* __restrict__ x,
                           const float* __restrict__ Ws, const float* __restrict__ bs,
                           const float* __restrict__ Wc, const float* __restrict__ bc,
                           float* __restrict__ hc, float* __restrict__ inp) {
  int idx = blockIdx.x * blockDim.x + threadIdx.x;   // b*512 + j
  int b = idx >> 9, j = idx & 511;
  const float* xl = x + ((long long)b * WIN_ + (WIN_ - 1)) * D_;
  float s1 = bs[j], s2 = bc[j];
  for (int d = 0; d < D_; ++d) {
    float xv = xl[d];
    s1 += xv * Ws[j * D_ + d];
    s2 += xv * Wc[j * D_ + d];
  }
  float h0 = tanhf(s1);
  hc[(long long)b * (2 * H_) + j]      = h0;
  hc[(long long)b * (2 * H_) + H_ + j] = sigmoidf_(s2);
  inp[(long long)b * KIN_ + (O_ + D_) + j] = h0;   // h columns
  if (j < O_) inp[(long long)b * KIN_ + j] = 0.0f; // y0 = 0
}

// ---------------------------------------------------------------------------
// Per-step kernels
// ---------------------------------------------------------------------------
// score[b,w] = sum_v V[v]*tanh(Wq[b,v] + Ukt[b,w,v]); softmax over w.
__global__ __launch_bounds__(256)
void attn_kernel(const float* __restrict__ Wq, const float* __restrict__ Ukt,
                 const float* __restrict__ V, float* __restrict__ att,
                 float* __restrict__ wout, int p) {
  __shared__ float sV[D_];
  __shared__ float s[WIN_];
  __shared__ float red[WIN_];
  const int b = blockIdx.x, t = threadIdx.x, lane = t & 31, wv = t >> 5;
  sV[t] = V[t];
  __syncthreads();
  const float* wq = Wq + (long long)b * D_;
  for (int w = wv; w < WIN_; w += 8) {
    const float* uk = Ukt + ((long long)b * WIN_ + w) * D_;
    float partial = 0.0f;
#pragma unroll
    for (int v = lane; v < D_; v += 32)
      partial += sV[v] * tanhf(wq[v] + uk[v]);
    for (int off = 16; off; off >>= 1) partial += __shfl_xor(partial, off, 32);
    if (lane == 0) s[w] = partial;
  }
  __syncthreads();
  float val = (t < WIN_) ? s[t] : -1e30f;
  if (t < WIN_) red[t] = val;
  __syncthreads();
  for (int off = 64; off >= 1; off >>= 1) {
    if (t < off && t + off < WIN_) red[t] = fmaxf(red[t], red[t + off]);
    __syncthreads();
  }
  const float mx = red[0];
  __syncthreads();
  float e = (t < WIN_) ? expf(val - mx) : 0.0f;
  if (t < WIN_) red[t] = e;
  __syncthreads();
  for (int off = 64; off >= 1; off >>= 1) {
    if (t < off && t + off < WIN_) red[t] += red[t + off];
    __syncthreads();
  }
  const float inv = 1.0f / red[0];
  if (t < WIN_) {
    float a = e * inv;
    att[(long long)b * WIN_ + t] = a;
    wout[((long long)b * P_ + p) * WIN_ + t] = a;
  }
}

// ctx[b,d] = sum_w att[b,w]*x[b,w,d]  -> inp[:, O_:O_+D_]
__global__ __launch_bounds__(256)
void ctx_kernel(const float* __restrict__ att, const float* __restrict__ x,
                float* __restrict__ inp) {
  __shared__ float sa[WIN_];
  const int b = blockIdx.x, t = threadIdx.x;
  if (t < WIN_) sa[t] = att[(long long)b * WIN_ + t];
  __syncthreads();
  float acc = 0.0f;
  const float* xb = x + (long long)b * WIN_ * D_ + t;
#pragma unroll 4
  for (int w = 0; w < WIN_; ++w) acc += sa[w] * xb[(long long)w * D_];
  inp[(long long)b * KIN_ + O_ + t] = acc;
}

// LSTM pointwise cell: consumes gates [i|f|g|o], updates hc, seeds inp h-cols.
__global__ void lstm_cell(const float* __restrict__ gates, float* __restrict__ hc,
                          float* __restrict__ inp) {
  int idx = blockIdx.x * blockDim.x + threadIdx.x;   // b*512 + j
  int b = idx >> 9, j = idx & 511;
  const float* g = gates + (long long)b * (4 * H_);
  float i_ = sigmoidf_(g[j]);
  float f_ = sigmoidf_(g[H_ + j]);
  float g_ = tanhf(g[2 * H_ + j]);
  float o_ = sigmoidf_(g[3 * H_ + j]);
  float c  = hc[(long long)b * (2 * H_) + H_ + j];
  float cn = f_ * c + i_ * g_;
  float hn = o_ * tanhf(cn);
  hc[(long long)b * (2 * H_) + j]      = hn;
  hc[(long long)b * (2 * H_) + H_ + j] = cn;
  inp[(long long)b * KIN_ + (O_ + D_) + j] = hn;
}

// ---------------------------------------------------------------------------
extern "C" void kernel_launch(void* const* d_in, const int* in_sizes, int n_in,
                              void* d_out, int out_size, void* d_ws, size_t ws_size,
                              hipStream_t stream) {
  (void)in_sizes; (void)n_in; (void)out_size; (void)ws_size;
  const float* x    = (const float*)d_in[0];
  const float* V    = (const float*)d_in[1];
  const float* W    = (const float*)d_in[2];
  const float* U    = (const float*)d_in[3];
  const float* W_ih = (const float*)d_in[4];
  const float* W_hh = (const float*)d_in[5];
  const float* b_ih = (const float*)d_in[6];
  const float* b_hh = (const float*)d_in[7];
  const float* Wd   = (const float*)d_in[8];
  const float* bd   = (const float*)d_in[9];
  const float* Ws   = (const float*)d_in[10];
  const float* bs   = (const float*)d_in[11];
  const float* Wc   = (const float*)d_in[12];
  const float* bc   = (const float*)d_in[13];

  float* out_y = (float*)d_out;                          // (B, P, O)
  float* out_w = out_y + (long long)B_ * P_ * O_;        // (B, P, WIN)

  // carve workspace (256B aligned)
  char* wp = (char*)d_ws;
  auto carve = [&](size_t bytes) -> char* {
    char* pp = wp; wp += (bytes + 255) & ~(size_t)255; return pp;
  };
  __bf16* U_bf  = (__bf16*)carve((size_t)D_ * D_ * 2);
  __bf16* W_bf  = (__bf16*)carve((size_t)D_ * 2 * H_ * 2);
  __bf16* Wcb   = (__bf16*)carve((size_t)4 * H_ * KIN_ * 2);
  __bf16* Wd_bf = (__bf16*)carve((size_t)O_ * H_ * 2);
  float*  Ukt   = (float*)carve((size_t)B_ * WIN_ * D_ * 4);  // (B, WIN, D) f32, L2-resident
  float*  hc    = (float*)carve((size_t)B_ * 2 * H_ * 4);     // [h | c]
  float*  inp   = (float*)carve((size_t)B_ * KIN_ * 4);       // [y | ctx | h]
  float*  Wqb   = (float*)carve((size_t)B_ * D_ * 4);
  float*  attb  = (float*)carve((size_t)B_ * WIN_ * 4);
  float*  gates = (float*)carve((size_t)B_ * 4 * H_ * 4);
  float*  bsum  = (float*)carve((size_t)4 * H_ * 4);

  auto gemm = [&](const float* A, int lda, long long sA,
                  const __bf16* Bw, int ldb,
                  float* C, int ldc, long long sC,
                  float* C2, int ldc2, const float* bias,
                  int M, int N, int K, int batch) {
    dim3 g(N / 64, M / 128, batch);
    gemm_bf16_tn<<<g, 256, 0, stream>>>(A, lda, sA, Bw, ldb, C, ldc, sC, C2, ldc2, bias, K);
  };

  // ---- one-time: weight conversion, bias combine, state init, Uk precompute
  cvt_bf16<<<(D_ * D_ + 255) / 256, 256, 0, stream>>>(U, U_bf, D_ * D_);
  cvt_bf16<<<(D_ * 2 * H_ + 255) / 256, 256, 0, stream>>>(W, W_bf, D_ * 2 * H_);
  cvt_bf16<<<(O_ * H_ + 255) / 256, 256, 0, stream>>>(Wd, Wd_bf, O_ * H_);
  build_wcomb<<<(4 * H_ * KIN_ + 255) / 256, 256, 0, stream>>>(W_ih, W_hh, Wcb);
  vec_add<<<(4 * H_ + 255) / 256, 256, 0, stream>>>(b_ih, b_hh, bsum, 4 * H_);
  init_state<<<B_ * H_ / 256, 256, 0, stream>>>(x, Ws, bs, Wc, bc, hc, inp);

  // Ukt[b,w,v] = sum_d x[b,w,d] * U[v,d]  (batched WMMA GEMM)
  gemm(x, D_, (long long)WIN_ * D_, U_bf, D_,
       Ukt, D_, (long long)WIN_ * D_, nullptr, 0, nullptr,
       WIN_, D_, D_, B_);

  // ---- sequential decode: 64 steps, 6 launches each (graph-capture safe)
  for (int pstep = 0; pstep < P_; ++pstep) {
    // Wq = [h|c] @ W^T        (512 x 256 x 1024)
    gemm(hc, 2 * H_, 0, W_bf, 2 * H_, Wqb, D_, 0, nullptr, 0, nullptr,
         B_, D_, 2 * H_, 1);
    // attention score + softmax (writes att and output weights)
    attn_kernel<<<B_, 256, 0, stream>>>(Wqb, Ukt, V, attb, out_w, pstep);
    // ctx = att @ x  -> inp ctx-columns
    ctx_kernel<<<B_, 256, 0, stream>>>(attb, x, inp);
    // gates = [y|ctx|h] @ [W_ih|W_hh]^T + (b_ih+b_hh)   (512 x 2048 x 832)
    gemm(inp, KIN_, 0, Wcb, KIN_, gates, 4 * H_, 0, nullptr, 0, bsum,
         B_, 4 * H_, KIN_, 1);
    // LSTM cell pointwise -> new h,c (and h columns of inp)
    lstm_cell<<<B_ * H_ / 256, 256, 0, stream>>>(gates, hc, inp);
    // y = h_new @ Wd^T + bd -> output slab (ldc = P*O) AND y-columns of inp
    gemm(hc, 2 * H_, 0, Wd_bf, H_, out_y + pstep * O_, P_ * O_, 0,
         inp, KIN_, bd, B_, O_, H_, 1);
  }
}
